// Attention_88837103550546
// MI455X (gfx1250) — matrix operand order repaired and checked by reading, not compiled, not run on hardware
//
#include <hip/hip_runtime.h>
#include <hip/hip_bf16.h>

// ---------------------------------------------------------------------------
// MI455X (gfx1250) attention forward. All three matmuls on
// v_wmma_f32_16x16x32_bf16 (f32 accumulate). One-time bf16 conversion /
// weight transposition; register-double-buffered GEMM K-loops; flash
// attention with double-buffered LDS strips, K staged via the CDNA5 async
// global->LDS engine (ASYNCcnt).
// B=2, N=2048, DIM=1024, HEADS=16, DIM_HEAD=64, q pre-scaled by sqrt(64)=8.
// ---------------------------------------------------------------------------

#define B_    2
#define N_    2048
#define D_    1024
#define H_    16
#define DH_   64
#define QKVC_ (3 * H_ * DH_)   // 3072
#define M_    (B_ * N_)        // 4096 rows of x

typedef __bf16    v16bf  __attribute__((ext_vector_type(16)));
typedef float     v8f    __attribute__((ext_vector_type(8)));
typedef unsigned  uint4v __attribute__((ext_vector_type(4)));
typedef float     f4v    __attribute__((ext_vector_type(4)));

struct bfrag { uint4v lo; uint4v hi; };   // 32B == v16bf

__device__ __forceinline__ unsigned short bf16bits(float f) {
    unsigned u = __builtin_bit_cast(unsigned, f);
    unsigned r = u + 0x7FFFu + ((u >> 16) & 1u);     // RNE
    return (unsigned short)(r >> 16);
}
__device__ __forceinline__ __bf16 f2bf(float f) {
    return __builtin_bit_cast(__bf16, bf16bits(f));
}

// Fragment from two 16B chunks (A and B both contiguous: K = {ka..ka+15}
// i.e. chunks at p and p+8).
__device__ __forceinline__ v16bf frag_ld2(const __bf16* p0, const __bf16* p1) {
    bfrag f;
    f.lo = *(const uint4v*)p0;
    f.hi = *(const uint4v*)p1;
    return __builtin_bit_cast(v16bf, f);
}

__device__ __forceinline__ v8f wmma_bf16(v16bf a, v16bf b, v8f c) {
    return __builtin_amdgcn_wmma_f32_16x16x32_bf16(
        false, a, false, b, (short)0, c, false, false);
}

// ---------------------------------------------------------------------------
// Prep 1: x f32 -> bf16 (same layout). 8 elements/thread, b128 in / b128 out.
// ---------------------------------------------------------------------------
__global__ __launch_bounds__(256) void convert_x_kernel(
    const float* __restrict__ in, __bf16* __restrict__ outb)
{
    const size_t i = ((size_t)blockIdx.x * 256 + threadIdx.x) * 8;
    f4v a = *(const f4v*)(in + i);
    f4v b = *(const f4v*)(in + i + 4);
    uint4v q;
    q.x = (unsigned)bf16bits(a.x) | ((unsigned)bf16bits(a.y) << 16);
    q.y = (unsigned)bf16bits(a.z) | ((unsigned)bf16bits(a.w) << 16);
    q.z = (unsigned)bf16bits(b.x) | ((unsigned)bf16bits(b.y) << 16);
    q.w = (unsigned)bf16bits(b.z) | ((unsigned)bf16bits(b.w) << 16);
    *(uint4v*)(outb + i) = q;
}

// ---------------------------------------------------------------------------
// Prep 2: transpose + convert weights: in f32 [K][C] -> out bf16 [C][K].
// ---------------------------------------------------------------------------
__global__ __launch_bounds__(256) void transpose_w_kernel(
    const float* __restrict__ in, __bf16* __restrict__ outb, int K, int C)
{
    __shared__ float tile[32][33];
    const int ctiles = C >> 5;
    const int ct = blockIdx.x % ctiles, kt = blockIdx.x / ctiles;
    const int c0 = ct * 32, k0 = kt * 32;
    const int tx = threadIdx.x & 31, ty = threadIdx.x >> 5;   // 32 x 8
    #pragma unroll
    for (int i = 0; i < 32; i += 8)
        tile[ty + i][tx] = in[(size_t)(k0 + ty + i) * C + c0 + tx];
    __syncthreads();
    #pragma unroll
    for (int i = 0; i < 32; i += 8)
        outb[(size_t)(c0 + ty + i) * K + k0 + tx] = f2bf(tile[tx][ty + i]);
}

// ---------------------------------------------------------------------------
// Kernel 1: qkv = xb @ wqkvT ; scatter to q/k/v [b,h,n,d] bf16, q *= 8.
// 16x64 tile per wave, register double-buffered K-loop (loads for step k+1
// issued before the WMMAs of step k -> partial loadcnt waits, latency hidden).
// ---------------------------------------------------------------------------
__global__ __launch_bounds__(256) void qkv_gemm_kernel(
    const __bf16* __restrict__ xb, const __bf16* __restrict__ wT,
    __bf16* __restrict__ qbuf, __bf16* __restrict__ kbuf, __bf16* __restrict__ vbuf)
{
    const int lane = threadIdx.x & 31;
    const int wave = threadIdx.x >> 5;
    const int lc   = lane & 15;
    const int half = lane >> 4;

    const int nT64   = QKVC_ / 64;               // 48
    const int mBlock = blockIdx.x / nT64;
    const int nTile  = blockIdx.x % nT64;
    const int mTile  = mBlock * 8 + wave;        // 0..255
    const int row    = mTile * 16 + lc;
    const int colBase = nTile * 64;

    const __bf16* ar = xb + (size_t)row * D_;
    const __bf16* b0 = wT + (size_t)(colBase      + lc) * D_;
    const __bf16* b1 = wT + (size_t)(colBase + 16 + lc) * D_;
    const __bf16* b2 = wT + (size_t)(colBase + 32 + lc) * D_;
    const __bf16* b3 = wT + (size_t)(colBase + 48 + lc) * D_;

    const int ka0 = half * 16, kb0 = half * 16;
    v16bf a_c  = frag_ld2(ar + ka0, ar + ka0 + 8);
    v16bf b0_c = frag_ld2(b0 + kb0, b0 + kb0 + 8);
    v16bf b1_c = frag_ld2(b1 + kb0, b1 + kb0 + 8);
    v16bf b2_c = frag_ld2(b2 + kb0, b2 + kb0 + 8);
    v16bf b3_c = frag_ld2(b3 + kb0, b3 + kb0 + 8);

    v8f acc0 = {}, acc1 = {}, acc2 = {}, acc3 = {};
    #pragma unroll 2
    for (int kk = 0; kk < D_; kk += 32) {
        const int kn  = (kk + 32) & (D_ - 1);     // wraps: last prefetch is a dummy
        const int kaN = kn + half * 16;
        const int kbN = kn + half * 16;
        const v16bf a_n  = frag_ld2(ar + kaN, ar + kaN + 8);
        const v16bf b0_n = frag_ld2(b0 + kbN, b0 + kbN + 8);
        const v16bf b1_n = frag_ld2(b1 + kbN, b1 + kbN + 8);
        const v16bf b2_n = frag_ld2(b2 + kbN, b2 + kbN + 8);
        const v16bf b3_n = frag_ld2(b3 + kbN, b3 + kbN + 8);
        acc0 = wmma_bf16(a_c, b0_c, acc0);
        acc1 = wmma_bf16(a_c, b1_c, acc1);
        acc2 = wmma_bf16(a_c, b2_c, acc2);
        acc3 = wmma_bf16(a_c, b3_c, acc3);
        a_c = a_n; b0_c = b0_n; b1_c = b1_n; b2_c = b2_n; b3_c = b3_n;
    }

    // 64-col tile lies inside one (qkv, head); d = t*16 + lc
    const int qkv = colBase / (H_ * DH_);
    const int h   = (colBase % (H_ * DH_)) / DH_;
    __bf16* dstb = (qkv == 0) ? qbuf : (qkv == 1) ? kbuf : vbuf;
    const float scale = (qkv == 0) ? 8.0f : 1.0f;   // fold q * sqrt(DH)
    #pragma unroll
    for (int v = 0; v < 8; ++v) {
        const int r  = mTile * 16 + v + half * 8;
        const int bb = r >> 11, n = r & (N_ - 1);
        __bf16* dst = dstb + ((size_t)(bb * H_ + h) * N_ + n) * DH_ + lc;
        dst[0]  = f2bf(acc0[v] * scale);
        dst[16] = f2bf(acc1[v] * scale);
        dst[32] = f2bf(acc2[v] * scale);
        dst[48] = f2bf(acc3[v] * scale);
    }
}

// ---------------------------------------------------------------------------
// Kernel 2: flash attention per (b,h). Block = 8 waves x 16 query rows.
// Double-buffered LDS strips: stage strip j+1 (K via async global->LDS,
// V transposed through VGPRs) while computing on strip j.
// ---------------------------------------------------------------------------
__global__ __launch_bounds__(256) void flash_attn_kernel(
    const __bf16* __restrict__ qbuf, const __bf16* __restrict__ kbuf,
    const __bf16* __restrict__ vbuf, const unsigned char* __restrict__ mask,
    __bf16* __restrict__ attnb)
{
    __shared__ __bf16 Ks[2][32][64];        // key strips, row-major (2 x 4 KB)
    __shared__ __bf16 Vt[2][64][32];        // value strips, transposed (2 x 4 KB)
    __shared__ __bf16 pStage[8][16][32];    // per-wave P staging (8 KB)

    const int lane = threadIdx.x & 31;
    const int wave = threadIdx.x >> 5;
    const int lc   = lane & 15;
    const int half = lane >> 4;

    const int bh    = blockIdx.x >> 4;               // 0..31
    const int qTile = (blockIdx.x & 15) * 8 + wave;  // 0..127
    const int b = bh / H_, h = bh % H_;
    const __bf16* Q = qbuf + (size_t)bh * N_ * DH_;
    const __bf16* K = kbuf + (size_t)bh * N_ * DH_;
    const __bf16* V = vbuf + (size_t)bh * N_ * DH_;
    const unsigned char* mrow = mask + (size_t)b * N_;

    // Q A-fragments (d 0..31 / 32..63), loaded once
    const __bf16* qr = Q + (size_t)(qTile * 16 + lc) * DH_;
    const int ka = half * 16;
    const v16bf aq0 = frag_ld2(qr + ka,      qr + ka + 8);
    const v16bf aq1 = frag_ld2(qr + 32 + ka, qr + 32 + ka + 8);

    float mrun[8], lsum[8], alpha[8];
    v8f o0 = {}, o1 = {}, o2 = {}, o3 = {};
    #pragma unroll
    for (int v = 0; v < 8; ++v) { mrun[v] = -3.402823466e38f; lsum[v] = 0.f; }

    const int sr = threadIdx.x >> 3;          // 0..31 staging row (key)
    const int sc = (threadIdx.x & 7) * 8;     // 0..56 staging d-chunk

    // --- stage strip 0 into buffer 0 ---
    {
        const unsigned ldsK = (unsigned)(size_t)&Ks[0][sr][sc];
        const unsigned long long gK =
            (unsigned long long)(size_t)(K + (size_t)sr * DH_ + sc);
        asm volatile("global_load_async_to_lds_b128 %0, %1, off"
                     :: "v"(ldsK), "v"(gK) : "memory");
        const __bf16* vp = V + (size_t)sr * DH_ + sc;
        #pragma unroll
        for (int i = 0; i < 8; ++i) Vt[0][sc + i][sr] = vp[i];
        asm volatile("s_wait_asynccnt 0" ::: "memory");
    }
    __syncthreads();

    for (int j0 = 0; j0 < N_; j0 += 32) {
        const int cur = (j0 >> 5) & 1;
        const int nxt = cur ^ 1;

        // stage next strip while computing on this one
        if (j0 + 32 < N_) {
            const unsigned ldsK = (unsigned)(size_t)&Ks[nxt][sr][sc];
            const unsigned long long gK =
                (unsigned long long)(size_t)(K + (size_t)(j0 + 32 + sr) * DH_ + sc);
            asm volatile("global_load_async_to_lds_b128 %0, %1, off"
                         :: "v"(ldsK), "v"(gK) : "memory");
            const __bf16* vp = V + (size_t)(j0 + 32 + sr) * DH_ + sc;
            #pragma unroll
            for (int i = 0; i < 8; ++i) Vt[nxt][sc + i][sr] = vp[i];
        }

        // S = Q K^T : two 16-key tiles x two 32-wide d chunks
        v8f s0 = {}, s1 = {};
        s0 = wmma_bf16(aq0, frag_ld2(&Ks[cur][lc][half * 16],      &Ks[cur][lc][half * 16 + 8]),      s0);
        s0 = wmma_bf16(aq1, frag_ld2(&Ks[cur][lc][32 + half * 16], &Ks[cur][lc][32 + half * 16 + 8]), s0);
        s1 = wmma_bf16(aq0, frag_ld2(&Ks[cur][16 + lc][half * 16],      &Ks[cur][16 + lc][half * 16 + 8]),      s1);
        s1 = wmma_bf16(aq1, frag_ld2(&Ks[cur][16 + lc][32 + half * 16], &Ks[cur][16 + lc][32 + half * 16 + 8]), s1);

        // key mask (per-lane column)
        const float NEG = -3.402823466e38f;
        const bool m0 = mrow[j0 + lc] != 0;
        const bool m1 = mrow[j0 + 16 + lc] != 0;
        #pragma unroll
        for (int v = 0; v < 8; ++v) {
            if (!m0) s0[v] = NEG;
            if (!m1) s1[v] = NEG;
        }

        // online softmax: each C-layout row lives on a 16-lane half
        #pragma unroll
        for (int v = 0; v < 8; ++v) {
            float mx = fmaxf(s0[v], s1[v]);
            #pragma unroll
            for (int off = 8; off >= 1; off >>= 1)
                mx = fmaxf(mx, __shfl_xor(mx, off, 16));
            const float mnew = fmaxf(mrun[v], mx);
            const float a    = __expf(mrun[v] - mnew);
            const float p0   = __expf(s0[v] - mnew);
            const float p1   = __expf(s1[v] - mnew);
            s0[v] = p0; s1[v] = p1;
            float rs = p0 + p1;
            #pragma unroll
            for (int off = 8; off >= 1; off >>= 1)
                rs += __shfl_xor(rs, off, 16);
            lsum[v]  = lsum[v] * a + rs;
            mrun[v]  = mnew;
            alpha[v] = a;
        }
        #pragma unroll
        for (int v = 0; v < 8; ++v) {
            o0[v] *= alpha[v]; o1[v] *= alpha[v];
            o2[v] *= alpha[v]; o3[v] *= alpha[v];
        }

        // P: C-layout f32 -> per-wave LDS bf16 -> A-layout reload
        #pragma unroll
        for (int v = 0; v < 8; ++v) {
            pStage[wave][v + half * 8][lc]      = f2bf(s0[v]);
            pStage[wave][v + half * 8][16 + lc] = f2bf(s1[v]);
        }
        asm volatile("s_wait_dscnt 0" ::: "memory");   // CDNA5 split DS counter
        const v16bf pA = frag_ld2(&pStage[wave][lc][ka], &pStage[wave][lc][ka + 8]);

        // O += P (16x32) x V (32x64); Vt rows are contiguous over keys
        o0 = wmma_bf16(pA, frag_ld2(&Vt[cur][lc][half * 16],      &Vt[cur][lc][half * 16 + 8]),      o0);
        o1 = wmma_bf16(pA, frag_ld2(&Vt[cur][16 + lc][half * 16], &Vt[cur][16 + lc][half * 16 + 8]), o1);
        o2 = wmma_bf16(pA, frag_ld2(&Vt[cur][32 + lc][half * 16], &Vt[cur][32 + lc][half * 16 + 8]), o2);
        o3 = wmma_bf16(pA, frag_ld2(&Vt[cur][48 + lc][half * 16], &Vt[cur][48 + lc][half * 16 + 8]), o3);

        // next strip's async K copy must be in LDS before anyone reads it
        asm volatile("s_wait_asynccnt 0" ::: "memory");
        __syncthreads();
    }

    // normalize and store to [b, n, h*64 + d] bf16
    #pragma unroll
    for (int v = 0; v < 8; ++v) {
        const float inv = 1.0f / lsum[v];
        const int n = qTile * 16 + v + half * 8;
        __bf16* dst = attnb + ((size_t)b * N_ + n) * (H_ * DH_) + h * DH_ + lc;
        dst[0]  = f2bf(o0[v] * inv);
        dst[16] = f2bf(o1[v] * inv);
        dst[32] = f2bf(o2[v] * inv);
        dst[48] = f2bf(o3[v] * inv);
    }
}

// ---------------------------------------------------------------------------
// Kernel 3: out = attn @ w_outT, f32 result. 16x64 tile per wave,
// register double-buffered K-loop.
// ---------------------------------------------------------------------------
__global__ __launch_bounds__(256) void out_gemm_kernel(
    const __bf16* __restrict__ attnb, const __bf16* __restrict__ wT,
    float* __restrict__ out)
{
    const int lane = threadIdx.x & 31;
    const int wave = threadIdx.x >> 5;
    const int lc   = lane & 15;
    const int half = lane >> 4;

    const int nT64   = D_ / 64;                  // 16
    const int mBlock = blockIdx.x / nT64;
    const int nTile  = blockIdx.x % nT64;
    const int mTile  = mBlock * 8 + wave;
    const int row    = mTile * 16 + lc;
    const int colBase = nTile * 64;

    const __bf16* ar = attnb + (size_t)row * D_;
    const __bf16* b0 = wT + (size_t)(colBase      + lc) * D_;
    const __bf16* b1 = wT + (size_t)(colBase + 16 + lc) * D_;
    const __bf16* b2 = wT + (size_t)(colBase + 32 + lc) * D_;
    const __bf16* b3 = wT + (size_t)(colBase + 48 + lc) * D_;

    const int ka0 = half * 16, kb0 = half * 16;
    v16bf a_c  = frag_ld2(ar + ka0, ar + ka0 + 8);
    v16bf b0_c = frag_ld2(b0 + kb0, b0 + kb0 + 8);
    v16bf b1_c = frag_ld2(b1 + kb0, b1 + kb0 + 8);
    v16bf b2_c = frag_ld2(b2 + kb0, b2 + kb0 + 8);
    v16bf b3_c = frag_ld2(b3 + kb0, b3 + kb0 + 8);

    v8f acc0 = {}, acc1 = {}, acc2 = {}, acc3 = {};
    #pragma unroll 2
    for (int kk = 0; kk < D_; kk += 32) {
        const int kn  = (kk + 32) & (D_ - 1);
        const int kaN = kn + half * 16;
        const int kbN = kn + half * 16;
        const v16bf a_n  = frag_ld2(ar + kaN, ar + kaN + 8);
        const v16bf b0_n = frag_ld2(b0 + kbN, b0 + kbN + 8);
        const v16bf b1_n = frag_ld2(b1 + kbN, b1 + kbN + 8);
        const v16bf b2_n = frag_ld2(b2 + kbN, b2 + kbN + 8);
        const v16bf b3_n = frag_ld2(b3 + kbN, b3 + kbN + 8);
        acc0 = wmma_bf16(a_c, b0_c, acc0);
        acc1 = wmma_bf16(a_c, b1_c, acc1);
        acc2 = wmma_bf16(a_c, b2_c, acc2);
        acc3 = wmma_bf16(a_c, b3_c, acc3);
        a_c = a_n; b0_c = b0_n; b1_c = b1_n; b2_c = b2_n; b3_c = b3_n;
    }
    #pragma unroll
    for (int v = 0; v < 8; ++v) {
        float* dst = out + (size_t)(mTile * 16 + v + half * 8) * D_ + colBase + lc;
        dst[0]  = acc0[v];
        dst[16] = acc1[v];
        dst[32] = acc2[v];
        dst[48] = acc3[v];
    }
}

// ---------------------------------------------------------------------------
extern "C" void kernel_launch(void* const* d_in, const int* in_sizes, int n_in,
                              void* d_out, int out_size, void* d_ws, size_t ws_size,
                              hipStream_t stream)
{
    const float*         x     = (const float*)d_in[0];
    const unsigned char* mask  = (const unsigned char*)d_in[1]; // jnp bool = 1B/elem
    const float*         w_qkv = (const float*)d_in[2];
    const float*         w_out = (const float*)d_in[3];
    float*               out   = (float*)d_out;

    // workspace (bf16): xb 8MB | wqkvT 6MB | woutT 2MB | q,k,v 24MB | attn 8MB
    __bf16* xb    = (__bf16*)d_ws;
    __bf16* wqkvT = xb    + (size_t)M_ * D_;
    __bf16* woutT = wqkvT + (size_t)QKVC_ * D_;
    __bf16* qbuf  = woutT + (size_t)D_ * D_;
    __bf16* kbuf  = qbuf  + (size_t)B_ * H_ * N_ * DH_;
    __bf16* vbuf  = kbuf  + (size_t)B_ * H_ * N_ * DH_;
    __bf16* attnb = vbuf  + (size_t)B_ * H_ * N_ * DH_;

    convert_x_kernel<<<dim3((M_ * D_) / (256 * 8)), dim3(256), 0, stream>>>(x, xb);
    transpose_w_kernel<<<dim3((D_ / 32) * (QKVC_ / 32)), dim3(256), 0, stream>>>(
        w_qkv, wqkvT, D_, QKVC_);
    transpose_w_kernel<<<dim3((D_ / 32) * (D_ / 32)), dim3(256), 0, stream>>>(
        w_out, woutT, D_, D_);

    qkv_gemm_kernel<<<dim3(32 * (QKVC_ / 64)), dim3(256), 0, stream>>>(
        xb, wqkvT, qbuf, kbuf, vbuf);
    flash_attn_kernel<<<dim3(B_ * H_ * 16), dim3(256), 0, stream>>>(
        qbuf, kbuf, vbuf, mask, attnb);
    out_gemm_kernel<<<dim3(32 * (D_ / 64)), dim3(256), 0, stream>>>(
        attnb, woutT, out);

    (void)in_sizes; (void)n_in; (void)out_size; (void)ws_size;
}
